// PointNetSetAbstractionMSG_46042049413723
// MI455X (gfx1250) — matrix operand order repaired and checked by reading, not compiled, not run on hardware
//
#include <hip/hip_runtime.h>
#include <hip/hip_bf16.h>

typedef __attribute__((ext_vector_type(16))) _Float16 v16h;
typedef __attribute__((ext_vector_type(8)))  _Float16 v8h;
typedef __attribute__((ext_vector_type(8)))  float    v8f;

#define BATCH   16
#define NPTS    4096
#define CIN     64
#define SPTS    1024
#define KMAX    64
#define BN_EPS  1e-5f

// ---------------------------------------------------------------------------
// FPS: one block per batch (32 waves). dist[] in LDS; per step: min-update
// scan, wave32 shuffle argmax, 32-entry LDS cross-wave argmax (2 barriers).
// ---------------------------------------------------------------------------
__global__ __launch_bounds__(1024)
void fps_kernel(const float* __restrict__ xyz, int* __restrict__ fps_idx,
                float* __restrict__ out_newxyz) {
  __shared__ float dist[NPTS];
  __shared__ float redv[32];
  __shared__ int   redi[32];
  __shared__ int   s_far;
  const int b = blockIdx.x;
  const int t = threadIdx.x;
  const float* base = xyz + (size_t)b * NPTS * 3;
  for (int i = t; i < NPTS; i += 1024) dist[i] = 1e10f;
  if (t == 0) s_far = 0;
  __syncthreads();
  for (int it = 0; it < SPTS; ++it) {
    const int far = s_far;
    const float cx = base[far * 3 + 0];
    const float cy = base[far * 3 + 1];
    const float cz = base[far * 3 + 2];
    if (t == 0) {
      fps_idx[b * SPTS + it] = far;
      out_newxyz[((size_t)b * SPTS + it) * 3 + 0] = cx;
      out_newxyz[((size_t)b * SPTS + it) * 3 + 1] = cy;
      out_newxyz[((size_t)b * SPTS + it) * 3 + 2] = cz;
    }
    float bv = -1.0f; int bi = 0;
    for (int i = t; i < NPTS; i += 1024) {
      float dx = base[i*3+0] - cx, dy = base[i*3+1] - cy, dz = base[i*3+2] - cz;
      float d  = dx*dx + dy*dy + dz*dz;
      float m  = fminf(dist[i], d);
      dist[i]  = m;
      if (m > bv) { bv = m; bi = i; }
    }
    #pragma unroll
    for (int d = 16; d > 0; d >>= 1) {
      float ov = __shfl_down(bv, d, 32);
      int   oi = __shfl_down(bi, d, 32);
      if (ov > bv) { bv = ov; bi = oi; }
    }
    if ((t & 31) == 0) { redv[t >> 5] = bv; redi[t >> 5] = bi; }
    __syncthreads();
    if (t < 32) {
      bv = redv[t]; bi = redi[t];
      #pragma unroll
      for (int d = 16; d > 0; d >>= 1) {
        float ov = __shfl_down(bv, d, 32);
        int   oi = __shfl_down(bi, d, 32);
        if (ov > bv) { bv = ov; bi = oi; }
      }
      if (t == 0) s_far = bi;
    }
    __syncthreads();
  }
}

// ---------------------------------------------------------------------------
// kNN: one block per (b,s), 8 waves. d2[] in LDS; 64 argmin extractions with
// wave32 shuffle reductions. Ascending order -> scales share prefixes.
// ---------------------------------------------------------------------------
__global__ __launch_bounds__(256)
void knn_kernel(const float* __restrict__ xyz, const float* __restrict__ newxyz,
                int* __restrict__ knn_idx) {
  __shared__ float d2[NPTS];
  __shared__ float redv[8];
  __shared__ int   redi[8];
  const int bs = blockIdx.x;
  const int b  = bs / SPTS;
  const int t  = threadIdx.x;
  const float cx = newxyz[(size_t)bs*3 + 0];
  const float cy = newxyz[(size_t)bs*3 + 1];
  const float cz = newxyz[(size_t)bs*3 + 2];
  const float* base = xyz + (size_t)b * NPTS * 3;
  for (int i = t; i < NPTS; i += 256) {
    float dx = base[i*3+0] - cx, dy = base[i*3+1] - cy, dz = base[i*3+2] - cz;
    d2[i] = dx*dx + dy*dy + dz*dz;
  }
  __syncthreads();
  for (int sel = 0; sel < KMAX; ++sel) {
    float bv = 3.0e38f; int bi = 0;
    for (int i = t; i < NPTS; i += 256) {
      float v = d2[i];
      if (v < bv) { bv = v; bi = i; }
    }
    #pragma unroll
    for (int d = 16; d > 0; d >>= 1) {
      float ov = __shfl_down(bv, d, 32);
      int   oi = __shfl_down(bi, d, 32);
      if (ov < bv) { bv = ov; bi = oi; }
    }
    if ((t & 31) == 0) { redv[t >> 5] = bv; redi[t >> 5] = bi; }
    __syncthreads();
    if (t < 32) {
      bv = (t < 8) ? redv[t] : 3.4e38f;
      bi = (t < 8) ? redi[t] : 0;
      #pragma unroll
      for (int d = 4; d > 0; d >>= 1) {
        float ov = __shfl_down(bv, d, 32);
        int   oi = __shfl_down(bi, d, 32);
        if (ov < bv) { bv = ov; bi = oi; }
      }
      if (t == 0) {
        knn_idx[(size_t)bs * KMAX + sel] = bi;
        d2[bi] = 3.9e38f;
      }
    }
    __syncthreads();
  }
}

// ---------------------------------------------------------------------------
// Gather: build f16 input [M, 96] = (xyz - center | features | zero pad),
// staged in registers, stored as 12 x b128.
// ---------------------------------------------------------------------------
__global__ __launch_bounds__(256)
void gather_kernel(const float* __restrict__ xyz, const float* __restrict__ feats,
                   const float* __restrict__ newxyz, const int* __restrict__ knn_idx,
                   _Float16* __restrict__ A, int k) {
  const int r  = blockIdx.x * 256 + threadIdx.x;     // row in [0, B*S*k)
  const int j  = r % k;
  const int bs = r / k;
  const int b  = bs / SPTS;
  const int id = knn_idx[(size_t)bs * KMAX + j];
  const float*  p  = xyz + ((size_t)b * NPTS + id) * 3;
  const float*  nc = newxyz + (size_t)bs * 3;
  const float4* f4 = (const float4*)(feats + ((size_t)b * NPTS + id) * CIN);
  _Float16 row[96];
  row[0] = (_Float16)(p[0] - nc[0]);
  row[1] = (_Float16)(p[1] - nc[1]);
  row[2] = (_Float16)(p[2] - nc[2]);
  #pragma unroll
  for (int q = 0; q < 16; ++q) {
    float4 v = f4[q];
    row[3 + q*4 + 0] = (_Float16)v.x;
    row[3 + q*4 + 1] = (_Float16)v.y;
    row[3 + q*4 + 2] = (_Float16)v.z;
    row[3 + q*4 + 3] = (_Float16)v.w;
  }
  #pragma unroll
  for (int c = 67; c < 96; ++c) row[c] = (_Float16)0.0f;
  v8h* dst = (v8h*)(A + (size_t)r * 96);
  #pragma unroll
  for (int q = 0; q < 12; ++q) {
    v8h o;
    #pragma unroll
    for (int i = 0; i < 8; ++i) o[i] = row[q*8 + i];
    dst[q] = o;
  }
}

// ---------------------------------------------------------------------------
// Convert weights to f16 [Cout][Kpad] (B-matrix source), K zero-padded.
// Per-scale block of 18432 halves: L1 64x96, L2 64x64, L3 128x64.
// ---------------------------------------------------------------------------
__global__ void wconv_kernel(const float* __restrict__ w0, const float* __restrict__ w1,
                             const float* __restrict__ w2, _Float16* __restrict__ wf) {
  const int t = blockIdx.x * blockDim.x + threadIdx.x;
  if (t >= 3 * 18432) return;
  const int s = t / 18432;
  const int r = t % 18432;
  _Float16* dst = wf + (size_t)s * 18432;
  if (r < 6144) {                       // L1: [64][96] from [64][67]
    int o = r / 96, kk = r % 96;
    float v = (kk < 67) ? w0[((size_t)s * 64 + o) * 67 + kk] : 0.0f;
    dst[r] = (_Float16)v;
  } else if (r < 10240) {               // L2: [64][64]
    int q = r - 6144; int o = q / 64, kk = q % 64;
    dst[r] = (_Float16)w1[((size_t)s * 64 + o) * 64 + kk];
  } else {                              // L3: [128][64]
    int q = r - 10240; int o = q / 64, kk = q % 64;
    dst[r] = (_Float16)w2[((size_t)s * 128 + o) * 64 + kk];
  }
}

// ---------------------------------------------------------------------------
// WMMA GEMM, register-blocked over N: each wave owns one 16-row M tile, loads
// its A fragments ONCE, then sweeps all Cout column tiles (W is L2-hot).
// 8 waves / block. Fully unrolled: NK*NT v_wmma_f32_16x16x32_f16 per wave.
// A/B fragments: contiguous K (16*half + s) in both; C/D layout per CDNA5 ISA 7.12.2.
// ---------------------------------------------------------------------------
template<int KPAD, int COUT>
__global__ __launch_bounds__(256)
void gemm_kernel(const _Float16* __restrict__ A, const _Float16* __restrict__ W,
                 _Float16* __restrict__ Out) {
  constexpr int NK = KPAD / 32;
  constexpr int NT = COUT / 16;
  const int lane  = threadIdx.x & 31;
  const int wv    = threadIdx.x >> 5;
  const int mrow0 = (blockIdx.x * 8 + wv) * 16;
  const int half  = lane >> 4;          // lane group 0..15 / 16..31
  const int l15   = lane & 15;
  // A fragments: row M = mrow0+l15; K = 16*half .. +15 (contiguous)
  const _Float16* arow = A + (size_t)(mrow0 + l15) * KPAD + half * 16;
  __builtin_prefetch(arow + (size_t)128 * KPAD, 0, 0);  // next M-super-tile
  v16h afrag[NK];
  #pragma unroll
  for (int kc = 0; kc < NK; ++kc) {
    v8h lo = *(const v8h*)(arow + kc * 32);
    v8h hi = *(const v8h*)(arow + kc * 32 + 8);
    #pragma unroll
    for (int i = 0; i < 8; ++i) { afrag[kc][i] = lo[i]; afrag[kc][8 + i] = hi[i]; }
  }
  #pragma unroll
  for (int nt = 0; nt < NT; ++nt) {
    v8f acc = {};
    #pragma unroll
    for (int kc = 0; kc < NK; ++kc) {
      // B fragment: col N = nt*16+l15; K = kc*32 + 16*half .. +15 (contiguous)
      v16h bfrag = *(const v16h*)(W + (size_t)(nt * 16 + l15) * KPAD + kc * 32 + half * 16);
      acc = __builtin_amdgcn_wmma_f32_16x16x32_f16(false, afrag[kc], false, bfrag,
                                                   (short)0, acc, false, false);
    }
    // C/D layout: VGPR r -> M = r + 8*half, N = l15
    _Float16* op = Out + (size_t)(mrow0 + half * 8) * COUT + nt * 16 + l15;
    #pragma unroll
    for (int r = 0; r < 8; ++r) op[(size_t)r * COUT] = (_Float16)acc[r];
  }
}

// ---------------------------------------------------------------------------
// BN statistics: per-channel sum / sumsq. Vectorized v8h reads; LDS float
// atomics (ds_add_f32) then global atomics.
// stats layout: [0..127]=sum  [128..255]=sumsq  [256..383]=mu  [384..511]=rsig
// ---------------------------------------------------------------------------
__global__ void zero_stats_kernel(float* __restrict__ stats) {
  int t = threadIdx.x;
  if (t < 512) stats[t] = 0.0f;
}

__global__ __launch_bounds__(256)
void stats_kernel(const _Float16* __restrict__ X, float* __restrict__ stats,
                  int Mrows, int Cch) {
  __shared__ float ls[128], lq[128];
  const int t = threadIdx.x;
  if (t < Cch) { ls[t] = 0.0f; lq[t] = 0.0f; }
  __syncthreads();
  const int nGrp    = Cch >> 3;                 // channel groups of 8
  const int cg      = t % nGrp;
  const int c0      = cg * 8;
  const int rowsPar = (gridDim.x * 256) / nGrp;
  const int rid     = (blockIdx.x * 256 + t) / nGrp;
  float s[8] = {}, q[8] = {};
  for (int r = rid; r < Mrows; r += rowsPar) {
    v8h v = *(const v8h*)(X + (size_t)r * Cch + c0);
    #pragma unroll
    for (int e = 0; e < 8; ++e) {
      float f = (float)v[e];
      s[e] += f; q[e] += f * f;
    }
  }
  #pragma unroll
  for (int e = 0; e < 8; ++e) {
    atomicAdd(&ls[c0 + e], s[e]);
    atomicAdd(&lq[c0 + e], q[e]);
  }
  __syncthreads();
  if (t < Cch) {
    atomicAdd(&stats[t], ls[t]);
    atomicAdd(&stats[128 + t], lq[t]);
  }
}

__global__ void finalize_stats_kernel(float* __restrict__ stats, int Cch, float invM) {
  const int t = threadIdx.x;
  if (t < Cch) {
    float mu  = stats[t] * invM;
    float var = stats[128 + t] * invM - mu * mu;
    stats[256 + t] = mu;
    stats[384 + t] = rsqrtf(var + BN_EPS);
  }
}

// ---------------------------------------------------------------------------
// BN + ReLU, 8 halves per thread (b128 in / b128 out).
// ---------------------------------------------------------------------------
__global__ __launch_bounds__(256)
void bnrelu_kernel(const _Float16* __restrict__ X, _Float16* __restrict__ Y,
                   const float* __restrict__ stats, const float* __restrict__ gamma,
                   const float* __restrict__ beta, int Cch, size_t totalVec) {
  size_t i = (size_t)blockIdx.x * 256 + threadIdx.x;
  if (i >= totalVec) return;
  const int c0 = (int)((i * 8) % Cch);          // Cch multiple of 8
  v8h x = ((const v8h*)X)[i];
  v8h y;
  #pragma unroll
  for (int e = 0; e < 8; ++e) {
    int c = c0 + e;
    float v = (float)x[e];
    float z = (v - stats[256 + c]) * stats[384 + c] * gamma[c] + beta[c];
    y[e] = (_Float16)fmaxf(z, 0.0f);
  }
  ((v8h*)Y)[i] = y;
}

// ---------------------------------------------------------------------------
// Layer-3 fused BN + ReLU + max over k neighbors -> d_out fused slice.
// ---------------------------------------------------------------------------
__global__ __launch_bounds__(256)
void bnmaxpool_kernel(const _Float16* __restrict__ X, float* __restrict__ outFused,
                      const float* __restrict__ stats, const float* __restrict__ gamma,
                      const float* __restrict__ beta, int k, int scaleOff) {
  const int t  = blockIdx.x * 256 + threadIdx.x;   // over B*S*128
  const int o  = t % 128;
  const int bs = t / 128;
  const float mu = stats[256 + o], rs = stats[384 + o];
  const float g = gamma[o], bt = beta[o];
  const _Float16* base = X + ((size_t)bs * k) * 128 + o;
  float m = 0.0f;   // relu outputs are >= 0
  for (int j = 0; j < k; ++j) {
    float v = ((float)base[(size_t)j * 128] - mu) * rs * g + bt;
    m = fmaxf(m, fmaxf(v, 0.0f));
  }
  outFused[(size_t)bs * 384 + scaleOff + o] = m;
}

// ---------------------------------------------------------------------------
extern "C" void kernel_launch(void* const* d_in, const int* in_sizes, int n_in,
                              void* d_out, int out_size, void* d_ws, size_t ws_size,
                              hipStream_t stream) {
  (void)in_sizes; (void)n_in; (void)out_size; (void)ws_size;
  const float* xyz   = (const float*)d_in[0];
  const float* feats = (const float*)d_in[1];
  const float* w0    = (const float*)d_in[2];
  const float* w1    = (const float*)d_in[3];
  const float* w2    = (const float*)d_in[4];
  const float* g0    = (const float*)d_in[5];
  const float* g1    = (const float*)d_in[6];
  const float* g2    = (const float*)d_in[7];
  const float* b0    = (const float*)d_in[8];
  const float* b1    = (const float*)d_in[9];
  const float* b2    = (const float*)d_in[10];

  float* out        = (float*)d_out;
  float* out_newxyz = out;                               // [B,S,3]
  float* out_fused  = out + (size_t)BATCH * SPTS * 3;    // [B,S,384]

  char* ws = (char*)d_ws;
  size_t off = 0;
  auto carve = [&](size_t bytes) -> char* {
    char* p = ws + off;
    off += (bytes + 255) & ~(size_t)255;
    return p;
  };
  int*      fps_idx = (int*)carve((size_t)BATCH * SPTS * 4);
  int*      knn_idx = (int*)carve((size_t)BATCH * SPTS * KMAX * 4);
  _Float16* wf      = (_Float16*)carve((size_t)3 * 18432 * 2);
  float*    stats   = (float*)carve(512 * 4);
  const size_t Mmax = (size_t)BATCH * SPTS * KMAX;
  _Float16* regA = (_Float16*)carve(Mmax * 96  * 2);     // gathered input
  _Float16* regB = (_Float16*)carve(Mmax * 128 * 2);     // pre-BN activations
  _Float16* regC = (_Float16*)carve(Mmax * 64  * 2);     // normalized activations

  fps_kernel<<<BATCH, 1024, 0, stream>>>(xyz, fps_idx, out_newxyz);
  knn_kernel<<<BATCH * SPTS, 256, 0, stream>>>(xyz, out_newxyz, knn_idx);
  wconv_kernel<<<(3 * 18432 + 255) / 256, 256, 0, stream>>>(w0, w1, w2, wf);

  for (int s = 0; s < 3; ++s) {
    const int    k    = 16 << s;
    const size_t M    = (size_t)BATCH * SPTS * k;
    const float  invM = 1.0f / (float)M;
    const _Float16* wL1 = wf + (size_t)s * 18432;
    const _Float16* wL2 = wL1 + 6144;
    const _Float16* wL3 = wL2 + 4096;
    const float *gm1 = g0 + s * 64,  *bt1 = b0 + s * 64;
    const float *gm2 = g1 + s * 64,  *bt2 = b1 + s * 64;
    const float *gm3 = g2 + s * 128, *bt3 = b2 + s * 128;

    gather_kernel<<<(unsigned)(M / 256), 256, 0, stream>>>(xyz, feats, out_newxyz,
                                                           knn_idx, regA, k);
    // Layer 1: [M,96] x [64,96]^T -> regB (pre-BN)
    zero_stats_kernel<<<1, 512, 0, stream>>>(stats);
    gemm_kernel<96, 64><<<(unsigned)(M / 128), 256, 0, stream>>>(regA, wL1, regB);
    stats_kernel<<<1024, 256, 0, stream>>>(regB, stats, (int)M, 64);
    finalize_stats_kernel<<<1, 128, 0, stream>>>(stats, 64, invM);
    bnrelu_kernel<<<(unsigned)((M * 64 / 8 + 255) / 256), 256, 0, stream>>>(
        regB, regC, stats, gm1, bt1, 64, M * 64 / 8);

    // Layer 2: [M,64] x [64,64]^T -> regB (pre-BN)
    zero_stats_kernel<<<1, 512, 0, stream>>>(stats);
    gemm_kernel<64, 64><<<(unsigned)(M / 128), 256, 0, stream>>>(regC, wL2, regB);
    stats_kernel<<<1024, 256, 0, stream>>>(regB, stats, (int)M, 64);
    finalize_stats_kernel<<<1, 128, 0, stream>>>(stats, 64, invM);
    bnrelu_kernel<<<(unsigned)((M * 64 / 8 + 255) / 256), 256, 0, stream>>>(
        regB, regC, stats, gm2, bt2, 64, M * 64 / 8);

    // Layer 3: [M,64] x [128,64]^T -> regB (pre-BN, 128ch), fused BN+max -> out
    zero_stats_kernel<<<1, 512, 0, stream>>>(stats);
    gemm_kernel<64, 128><<<(unsigned)(M / 128), 256, 0, stream>>>(regC, wL3, regB);
    stats_kernel<<<1024, 256, 0, stream>>>(regB, stats, (int)M, 128);
    finalize_stats_kernel<<<1, 128, 0, stream>>>(stats, 128, invM);
    bnmaxpool_kernel<<<(BATCH * SPTS * 128) / 256, 256, 0, stream>>>(
        regB, out_fused, stats, gm3, bt3, k, s * 128);
  }
}
